// NegativeGraphSampler_20890720927937
// MI455X (gfx1250) — 8-bit matrix operand order repaired and checked by reading, compiled, not run on hardware
//
#include <hip/hip_runtime.h>
#include <hip/hip_bf16.h>
#include <stdint.h>

typedef int v8i __attribute__((ext_vector_type(8)));
typedef unsigned int u32;

#define NNODES    4096
#define NEDGES    8192u
#define LOG2_E    13
#define EE        67108864u
#define HALF_EE   33554432u
#define KEEP_CAP  16384u
#define ROW_WORDS 128
#define SCAN_BLOCKS 1024
#define IPT 256u
#define IPB 65536u

#define WS_MASK 0u
#define WS_CNT  524288u
#define WS_MULT 528384u
#define WS_FORB 532480u
#define WS_BCNT 536576u
#define WS_BOFF 537600u
#define WS_SCAL 538624u
#define WS_CNT8 540672u
#define WS_ZERO_WORDS 537600u

__device__ __forceinline__ void async_b128_to_lds(u32 lds_off, const void* gptr) {
  asm volatile("global_load_async_to_lds_b128 %0, %1, off"
               :: "v"(lds_off), "v"((unsigned long long)(uintptr_t)gptr)
               : "memory");
}
__device__ __forceinline__ void wait_async0() {
  asm volatile("s_wait_asynccnt 0x0" ::: "memory");
}

__device__ __forceinline__ u32 rotl32(u32 x, int r) { return (x << r) | (x >> (32 - r)); }

__device__ __forceinline__ void threefry2x32_0_42(u32 c0, u32 c1, u32* o0, u32* o1) {
  const u32 k0 = 0u, k1 = 42u;
  const u32 k2 = k0 ^ k1 ^ 0x1BD11BDAu;
  u32 x0 = c0 + k0, x1 = c1 + k1;
#define TF_RND(r) { x0 += x1; x1 = rotl32(x1, (r)); x1 ^= x0; }
  TF_RND(13) TF_RND(15) TF_RND(26) TF_RND(6)  x0 += k1; x1 += k2 + 1u;
  TF_RND(17) TF_RND(29) TF_RND(16) TF_RND(24) x0 += k2; x1 += k0 + 2u;
  TF_RND(13) TF_RND(15) TF_RND(26) TF_RND(6)  x0 += k0; x1 += k1 + 3u;
  TF_RND(17) TF_RND(29) TF_RND(16) TF_RND(24) x0 += k1; x1 += k2 + 4u;
  TF_RND(13) TF_RND(15) TF_RND(26) TF_RND(6)  x0 += k2; x1 += k0 + 5u;
#undef TF_RND
  *o0 = x0; *o1 = x1;
}

__device__ __forceinline__ bool keep_staged(u32 t, const u32* maskL, const int* dstL,
                                            float kp) {
  u32 r = (t >> LOG2_E) & 7u;
  u32 j = t & (NEDGES - 1u);
  int d = dstL[j];
  u32 w = maskL[r * ROW_WORDS + ((u32)d >> 5)];
  bool m = ((w >> ((u32)d & 31u)) & 1u) == 0u;
  u32 o0, o1, bits;
  if (t < HALF_EE) { threefry2x32_0_42(t, t + HALF_EE, &o0, &o1); bits = o0; }
  else             { threefry2x32_0_42(t - HALF_EE, t, &o0, &o1); bits = o1; }
  float u = __uint_as_float((bits >> 9) | 0x3F800000u) - 1.0f;
  return m & (u < kp);
}

__global__ void ngs_init_ws(u32* __restrict__ ws) {
  for (u32 t = blockIdx.x * blockDim.x + threadIdx.x; t < WS_ZERO_WORDS;
       t += gridDim.x * blockDim.x)
    ws[t] = 0u;
}

__global__ void ngs_init_out(const int* __restrict__ esrc, const int* __restrict__ edst,
                             int* __restrict__ out) {
  u32 t = blockIdx.x * blockDim.x + threadIdx.x;
  if (t < KEEP_CAP)            out[t] = esrc[0];
  else if (t < 2u * KEEP_CAP)  out[t] = edst[0];
}

__global__ void ngs_build(const int* __restrict__ esrc, const int* __restrict__ edst,
                          u32* __restrict__ ws) {
  u32* mask = ws + WS_MASK;
  u32* cnt  = ws + WS_CNT;
  u32* mult = ws + WS_MULT;
  u32 e = blockIdx.x * blockDim.x + threadIdx.x;
  if (e >= NEDGES) return;
  int s = esrc[e], d = edst[e];
  atomicOr(&mask[(u32)s * ROW_WORDS + ((u32)d >> 5)], 1u << ((u32)d & 31u));
  atomicOr(&mask[(u32)s * ROW_WORDS + ((u32)s >> 5)], 1u << ((u32)s & 31u));
  atomicAdd(&cnt[d], 1u);
  atomicAdd(&mult[s], 1u);
}

__global__ void ngs_pack_cnt(const u32* __restrict__ cnt, u32* __restrict__ cnt8) {
  u32 w = blockIdx.x * blockDim.x + threadIdx.x;
  if (w >= 1024u) return;
  u32 lo = 0u, hi = 0u;
  for (u32 q = 0; q < 4u; ++q) {
    u32 c = cnt[w * 4u + q];
    lo |= (c & 0xFFu) << (8u * q);
    hi |= ((c >> 8) & 0xFFu) << (8u * q);
  }
  cnt8[w] = lo;
  cnt8[1024u + w] = hi;
}

__global__ void ngs_forb_wmma(const u32* __restrict__ ws_mask,
                              const u32* __restrict__ cnt8,
                              u32* __restrict__ forb) {
  __shared__ int dmat[16 * 16];
  const u32 lane = threadIdx.x;
  const u32 m    = lane & 15u;
  const u32 half = lane >> 4;
  const u32 n0   = blockIdx.x * 16u;
  const u32 node = n0 + m;
  const u32 col  = (m < 2u) ? m : 0u;

  union { v8i v; int s[8]; } a, b, acc;
  for (int q = 0; q < 8; ++q) acc.s[q] = 0;

  for (u32 kc = 0; kc < (u32)NNODES; kc += 64u) {
    for (int v = 0; v < 8; ++v) {
      u32 offs = (((u32)v & 1u) << 2) | ((((u32)v >> 1) & 1u) << 4) |
                 (((u32)v >> 2) << 5) | (half << 3);
      u32 word = ws_mask[node * ROW_WORDS + ((kc + offs) >> 5)];
      u32 nib  = (word >> (offs & 31u)) & 0xFu;
      a.s[v] = (int)((nib & 1u) | (((nib >> 1) & 1u) << 8) |
                     (((nib >> 2) & 1u) << 16) | (((nib >> 3) & 1u) << 24));
    }
    for (int v = 0; v < 8; ++v) {
      u32 word = cnt8[col * 1024u + (kc >> 2) + (half << 1) + ((u32)v & 1u) + ((((u32)v >> 1) & 1u) << 2) + (((u32)v >> 2) << 3)];
      b.s[v] = (m < 2u) ? (int)word : 0;
    }
    acc.v = __builtin_amdgcn_wmma_i32_16x16x64_iu8(false, a.v, false, b.v,
                                                   acc.v, false, false);
  }
  for (int g = 0; g < 8; ++g)
    dmat[(g + (int)(half << 3)) * 16 + (int)m] = acc.s[g];
  __syncthreads();
  if (lane < 16u) {
    int f = dmat[(int)lane * 16 + 0] + (dmat[(int)lane * 16 + 1] << 8);
    forb[n0 + lane] = (u32)f;
  }
}

__global__ void ngs_numneg(const u32* __restrict__ mult, const u32* __restrict__ forb,
                           u32* __restrict__ scal) {
  __shared__ u32 red[256];
  u32 t = threadIdx.x, sum = 0u;
  for (u32 n = t; n < (u32)NNODES; n += 256u)
    sum += mult[n] * (NEDGES - forb[n]);
  red[t] = sum; __syncthreads();
  for (u32 s = 128u; s > 0u; s >>= 1) { if (t < s) red[t] += red[t + s]; __syncthreads(); }
  if (t == 0u) {
    u32 total = red[0];
    u32 ratio = total >> LOG2_E;
    float kp  = 2.0f / (float)ratio;
    scal[0] = total;
    ((float*)scal)[1] = kp;
  }
}

__global__ void ngs_pass1(const u32* __restrict__ mask, const int* __restrict__ esrc,
                          const int* __restrict__ edst, const u32* __restrict__ scal,
                          u32* __restrict__ bcnt) {
  __shared__ __align__(16) int lds_dst[NEDGES];
  __shared__ __align__(16) u32 lds_mask[8 * ROW_WORDS];
  __shared__ int lds_src[8];
  __shared__ u32 red[256];
  const u32 th = threadIdx.x;
  const u32 i0 = blockIdx.x * 8u;

  if (th < 8u) lds_src[th] = esrc[i0 + th];
  for (u32 q = 0; q < 8u; ++q) {
    u32 e4 = th + q * 256u;
    async_b128_to_lds((u32)(uintptr_t)&lds_dst[e4 * 4u], edst + e4 * 4u);
  }
  __syncthreads();
  {
    u32 r = th >> 5, c = th & 31u;
    const u32* rowp = mask + (u32)lds_src[r] * ROW_WORDS + c * 4u;
    async_b128_to_lds((u32)(uintptr_t)&lds_mask[r * ROW_WORDS + c * 4u], rowp);
  }
  wait_async0();
  __syncthreads();

  float kp = ((const float*)scal)[1];
  u32 base = blockIdx.x * IPB + th * IPT;
  u32 c = 0u;
  for (u32 q = 0; q < IPT; ++q)
    c += keep_staged(base + q, lds_mask, lds_dst, kp) ? 1u : 0u;
  red[th] = c; __syncthreads();
  for (u32 s = 128u; s > 0u; s >>= 1) { if (th < s) red[th] += red[th + s]; __syncthreads(); }
  if (th == 0u) bcnt[blockIdx.x] = red[0];
}

__global__ void ngs_scan(const u32* __restrict__ bcnt, u32* __restrict__ boff) {
  if (blockIdx.x == 0u && threadIdx.x == 0u) {
    u32 run = 0u;
    for (u32 b = 0; b < SCAN_BLOCKS; ++b) { boff[b] = run; run += bcnt[b]; }
  }
}

__global__ void ngs_pass2(const u32* __restrict__ mask, const int* __restrict__ esrc,
                          const int* __restrict__ edst, const u32* __restrict__ scal,
                          const u32* __restrict__ boff, int* __restrict__ out) {
  __shared__ __align__(16) int lds_dst[NEDGES];
  __shared__ __align__(16) u32 lds_mask[8 * ROW_WORDS];
  __shared__ int lds_src[8];
  __shared__ u32 cnts[256];
  u32 blockBase = boff[blockIdx.x];
  if (blockBase >= KEEP_CAP) return;
  const u32 th = threadIdx.x;
  const u32 i0 = blockIdx.x * 8u;

  if (th < 8u) lds_src[th] = esrc[i0 + th];
  for (u32 q = 0; q < 8u; ++q) {
    u32 e4 = th + q * 256u;
    async_b128_to_lds((u32)(uintptr_t)&lds_dst[e4 * 4u], edst + e4 * 4u);
  }
  __syncthreads();
  {
    u32 r = th >> 5, c = th & 31u;
    const u32* rowp = mask + (u32)lds_src[r] * ROW_WORDS + c * 4u;
    async_b128_to_lds((u32)(uintptr_t)&lds_mask[r * ROW_WORDS + c * 4u], rowp);
  }
  wait_async0();
  __syncthreads();

  float kp = ((const float*)scal)[1];
  u32 base = blockIdx.x * IPB + th * IPT;
  u32 c = 0u;
  for (u32 q = 0; q < IPT; ++q)
    c += keep_staged(base + q, lds_mask, lds_dst, kp) ? 1u : 0u;
  cnts[th] = c; __syncthreads();
  for (u32 off = 1u; off < 256u; off <<= 1) {
    u32 v = (th >= off) ? cnts[th - off] : 0u;
    __syncthreads();
    cnts[th] += v;
    __syncthreads();
  }
  u32 wr = blockBase + (cnts[th] - c);
  int* outSrc = out;
  int* outDst = out + KEEP_CAP;
  for (u32 q = 0; q < IPT; ++q) {
    u32 t = base + q;
    if (keep_staged(t, lds_mask, lds_dst, kp)) {
      if (wr < KEEP_CAP) {
        u32 r = (t >> LOG2_E) & 7u;
        u32 j = t & (NEDGES - 1u);
        outSrc[wr] = lds_src[r];
        outDst[wr] = lds_dst[j];
      }
      ++wr;
    }
  }
}

extern "C" void kernel_launch(void* const* d_in, const int* in_sizes, int n_in,
                              void* d_out, int out_size, void* d_ws, size_t ws_size,
                              hipStream_t stream) {
  (void)in_sizes; (void)n_in; (void)out_size; (void)ws_size;
  const int* esrc = (const int*)d_in[1];
  const int* edst = (const int*)d_in[2];
  int* out = (int*)d_out;
  u32* ws  = (u32*)d_ws;

  ngs_init_ws<<<1024, 256, 0, stream>>>(ws);
  ngs_init_out<<<(2u * KEEP_CAP + 255u) / 256u, 256, 0, stream>>>(esrc, edst, out);
  ngs_build<<<(NEDGES + 255u) / 256u, 256, 0, stream>>>(esrc, edst, ws);
  ngs_pack_cnt<<<4, 256, 0, stream>>>(ws + WS_CNT, ws + WS_CNT8);
  ngs_forb_wmma<<<NNODES / 16, 32, 0, stream>>>(ws + WS_MASK, ws + WS_CNT8, ws + WS_FORB);
  ngs_numneg<<<1, 256, 0, stream>>>(ws + WS_MULT, ws + WS_FORB, ws + WS_SCAL);
  ngs_pass1<<<SCAN_BLOCKS, 256, 0, stream>>>(ws + WS_MASK, esrc, edst,
                                             ws + WS_SCAL, ws + WS_BCNT);
  ngs_scan<<<1, 32, 0, stream>>>(ws + WS_BCNT, ws + WS_BOFF);
  ngs_pass2<<<SCAN_BLOCKS, 256, 0, stream>>>(ws + WS_MASK, esrc, edst,
                                             ws + WS_SCAL, ws + WS_BOFF, out);
}
